// MOE_28561532519116
// MI455X (gfx1250) — matrix operand order repaired and checked by reading, not compiled, not run on hardware
//
#include <hip/hip_runtime.h>
#include <hip/hip_bf16.h>

#define T_TOK 4096
#define DDIM  2048
#define NEXP  8
#define FDIM  512
#define LDS_PAD 8

typedef __attribute__((ext_vector_type(16))) __bf16 v16bf;
typedef __attribute__((ext_vector_type(8)))  __bf16 v8bf;
typedef __attribute__((ext_vector_type(8)))  float  v8f;

// ---------------------------------------------------------------------------
// Fragment loaders (CDNA5 WMMA 16x16x32 bf16, wave32).
// A 16x32 (ISA 7.12.2): lane row = lane&15; lanes 0-15: K = k..k+7, k+16..k+23;
// lanes 16-31: K = k+8..k+15, k+24..k+31.  -> two 16B loads per lane.
// B 32x16 from row-major weight [N,K]: lane col = lane&15; same K order as A:
// lanes 0-15: K = k..k+7, k+16..k+23; lanes 16-31: K = k+8..k+15, k+24..k+31.
// ---------------------------------------------------------------------------
__device__ __forceinline__ v16bf load_frag_a(const __bf16* rowp, int k) {
    v16bf f;
#pragma unroll
    for (int i = 0; i < 8; ++i) f[i] = rowp[k + i];
#pragma unroll
    for (int i = 0; i < 8; ++i) f[8 + i] = rowp[k + 16 + i];
    return f;
}

__device__ __forceinline__ v16bf load_frag_b(const __bf16* wrow, int k) {
    v16bf f;
#pragma unroll
    for (int i = 0; i < 8; ++i) f[i] = wrow[k + i];
#pragma unroll
    for (int i = 0; i < 8; ++i) f[8 + i] = wrow[k + 16 + i];
    return f;
}

__device__ __forceinline__ v8f wmma_bf16(v16bf a, v16bf b, v8f c) {
    return __builtin_amdgcn_wmma_f32_16x16x32_bf16(
        false, a, false, b, (short)0, c, false, false);
}

__device__ __forceinline__ float fast_sigmoid(float x) {
    return __builtin_amdgcn_rcpf(1.0f + __expf(-x));
}
__device__ __forceinline__ float fast_silu(float x) {
    return x * fast_sigmoid(x);
}

// ---------------------------------------------------------------------------
// Generic fp32 -> bf16 cast (vectorized x4); used for h and all weights.
// ---------------------------------------------------------------------------
__global__ __launch_bounds__(256) void cast_bf16_kernel(const float* __restrict__ src,
                                                        __bf16* __restrict__ dst) {
    int i = (blockIdx.x * blockDim.x + threadIdx.x) * 4;
    float4 v = *(const float4*)(src + i);
    __bf16* o = dst + i;
    o[0] = (__bf16)v.x; o[1] = (__bf16)v.y; o[2] = (__bf16)v.z; o[3] = (__bf16)v.w;
}

// ---------------------------------------------------------------------------
// Router: one wave per token: 8 expert logits + shared sigmoid gate.
// ---------------------------------------------------------------------------
__global__ __launch_bounds__(256) void router_kernel(
    const float* __restrict__ h, const float* __restrict__ gate_w,
    const float* __restrict__ sg_w, float* __restrict__ logits_out,
    float* __restrict__ sgate, float* __restrict__ rweight,
    int* __restrict__ buckets, int* __restrict__ counts) {
    int t    = blockIdx.x * 8 + (threadIdx.x >> 5);
    int lane = threadIdx.x & 31;
    const float* hp = h + (size_t)t * DDIM;

    float acc[9];
#pragma unroll
    for (int e = 0; e < 9; ++e) acc[e] = 0.f;

    for (int d = lane; d < DDIM; d += 32) {
        float hv = hp[d];
#pragma unroll
        for (int e = 0; e < NEXP; ++e) acc[e] += hv * gate_w[e * DDIM + d];
        acc[8] += hv * sg_w[d];
    }
#pragma unroll
    for (int e = 0; e < 9; ++e)
        for (int off = 16; off; off >>= 1) acc[e] += __shfl_xor(acc[e], off, 32);

    if (lane == 0) {
        float mx = acc[0];
#pragma unroll
        for (int e = 1; e < NEXP; ++e) mx = fmaxf(mx, acc[e]);
        float p[NEXP];
#pragma unroll
        for (int e = 0; e < NEXP; ++e) {
            logits_out[t * NEXP + e] = acc[e];
            p[e] = __expf(acc[e] - mx);
        }
        int i1 = 0; float p1 = p[0];
#pragma unroll
        for (int e = 1; e < NEXP; ++e) if (p[e] > p1) { p1 = p[e]; i1 = e; }
        int i2 = (i1 == 0) ? 1 : 0; float p2 = p[i2];
#pragma unroll
        for (int e = 0; e < NEXP; ++e)
            if (e != i1 && p[e] > p2) { p2 = p[e]; i2 = e; }
        float inv = __builtin_amdgcn_rcpf(p1 + p2);
        rweight[t * 2 + 0] = p1 * inv;
        rweight[t * 2 + 1] = p2 * inv;
        sgate[t] = fast_sigmoid(acc[8]);
        int pos1 = atomicAdd(&counts[i1], 1);
        buckets[i1 * T_TOK + pos1] = (t << 1);
        int pos2 = atomicAdd(&counts[i2], 1);
        buckets[i2 * T_TOK + pos2] = (t << 1) | 1;
    }
}

// ---------------------------------------------------------------------------
// Stage 1 (routed AND shared): dest[row] = silu(x@W0^T+b0)*(x@W1^T+b1)
// buckets==nullptr -> identity rows (shared expert), counts==nullptr -> T_TOK.
// Block = 64 tokens staged in LDS; 8 waves: wave owns 2 M-tiles (mh = w&1)
// and column group ng = w>>1; 4 WMMAs per K-step reuse both B fragments.
// Out-of-range rows are clamped duplicates of row count-1 -> their outputs and
// destinations are identical to that row, so stores need no guards.
// ---------------------------------------------------------------------------
__global__ __launch_bounds__(256) void mlp1_kernel(
    const __bf16* __restrict__ x16,
    const __bf16* __restrict__ w0b, const float* __restrict__ b0b,
    const __bf16* __restrict__ w1b, const float* __restrict__ b1b,
    const int* __restrict__ buckets, const int* __restrict__ counts,
    __bf16* __restrict__ dest) {
    extern __shared__ __bf16 lds[];
    const int LSTR = DDIM + LDS_PAD;
    int e = blockIdx.y;
    int count = counts ? counts[e] : T_TOK;
    int m0 = blockIdx.x * 64;
    if (m0 >= count) return;
    const int* bkt = buckets ? buckets + e * T_TOK : nullptr;

    // ---- stage A tile (64 x DDIM bf16) into LDS, 4 threads per row ----
    {
        int tid = threadIdx.x;
        int lrow = tid >> 2;
        int grow = m0 + lrow;
        int src_row;
        if (bkt) { int rc = grow < count ? grow : count - 1; src_row = bkt[rc] >> 1; }
        else     { src_row = grow; }
        const v8bf* gp = (const v8bf*)(x16 + (size_t)src_row * DDIM + (tid & 3) * (DDIM / 4));
        v8bf* lp = (v8bf*)(lds + lrow * LSTR + (tid & 3) * (DDIM / 4));
#pragma unroll 4
        for (int c = 0; c < DDIM / 4 / 8; ++c) lp[c] = gp[c];
    }
    __syncthreads();

    int wave = threadIdx.x >> 5, lane = threadIdx.x & 31;
    int mh = wave & 1, ng = wave >> 1;
    int laneHi = lane >> 4;
    int koff = laneHi << 3;
    int ncol = lane & 15;

    const __bf16* ap[2];
    __bf16* dp[2][8];   // nf-invariant destination row pointers (clamped)
#pragma unroll
    for (int j = 0; j < 2; ++j) {
        ap[j] = lds + ((mh * 2 + j) * 16 + (lane & 15)) * LSTR + (laneHi << 3);
#pragma unroll
        for (int r = 0; r < 8; ++r) {
            int gr = m0 + (mh * 2 + j) * 16 + r + (laneHi << 3);
            int grc = gr < count ? gr : count - 1;
            int ent = bkt ? bkt[grc] : grc;
            dp[j][r] = dest + (size_t)ent * FDIM;
        }
    }

    const __bf16* w0e = w0b + (size_t)e * FDIM * DDIM;
    const __bf16* w1e = w1b + (size_t)e * FDIM * DDIM;

    for (int nf = ng; nf < FDIM / 16; nf += 4) {
        int n = nf * 16 + ncol;
        const __bf16* g_row = w0e + (size_t)n * DDIM + koff;
        const __bf16* u_row = w1e + (size_t)n * DDIM + koff;
        v8f accG[2], accU[2];
#pragma unroll
        for (int j = 0; j < 2; ++j) {
            accG[j] = (v8f){0, 0, 0, 0, 0, 0, 0, 0};
            accU[j] = (v8f){0, 0, 0, 0, 0, 0, 0, 0};
        }
        for (int k = 0; k < DDIM; k += 32) {
            v16bf bg = load_frag_b(g_row, k);
            v16bf bu = load_frag_b(u_row, k);
#pragma unroll
            for (int j = 0; j < 2; ++j) {
                v16bf fa = load_frag_a(ap[j], k);
                accG[j] = wmma_bf16(fa, bg, accG[j]);
                accU[j] = wmma_bf16(fa, bu, accU[j]);
            }
        }
        float bg0 = b0b[e * FDIM + n], bu0 = b1b[e * FDIM + n];
#pragma unroll
        for (int j = 0; j < 2; ++j) {
#pragma unroll
            for (int r = 0; r < 8; ++r) {
                float g = accG[j][r] + bg0;
                float u = accU[j][r] + bu0;
                dp[j][r][n] = (__bf16)(fast_silu(g) * u);
            }
        }
    }
}

// ---------------------------------------------------------------------------
// Routed stage 2: atomicAdd( rw * (inter@wo^T + bo) ) onto out.
// Out-of-range rows get rw = 0 -> their atomicAdd contributes 0 (branch-free).
// ---------------------------------------------------------------------------
__global__ __launch_bounds__(256) void moe_mlp2_kernel(
    const __bf16* __restrict__ inter,
    const __bf16* __restrict__ wo, const float* __restrict__ bo,
    const int* __restrict__ buckets, const int* __restrict__ counts,
    const float* __restrict__ rweight, float* __restrict__ out) {
    extern __shared__ __bf16 lds[];
    const int LSTR = FDIM + LDS_PAD;
    int e = blockIdx.y;
    int count = counts[e];
    int m0 = blockIdx.x * 64;
    if (m0 >= count) return;
    const int* bkt = buckets + e * T_TOK;

    {   // stage 64 x FDIM bf16 gathered inter rows (row index = entry)
        int tid = threadIdx.x;
        int lrow = tid >> 2;
        int grow = m0 + lrow;
        int rc = grow < count ? grow : count - 1;
        int src_row = bkt[rc];
        const v8bf* gp = (const v8bf*)(inter + (size_t)src_row * FDIM + (tid & 3) * (FDIM / 4));
        v8bf* lp = (v8bf*)(lds + lrow * LSTR + (tid & 3) * (FDIM / 4));
#pragma unroll
        for (int c = 0; c < FDIM / 4 / 8; ++c) lp[c] = gp[c];
    }
    __syncthreads();

    int wave = threadIdx.x >> 5, lane = threadIdx.x & 31;
    int mh = wave & 1, ng = wave >> 1;
    int laneHi = lane >> 4;
    int koff = laneHi << 3;
    int ncol = lane & 15;

    const __bf16* ap[2];
    float* op[2][8];    // nf-invariant output row pointers
    float  rwv[2][8];   // routing weights (0 for clamped tail rows)
#pragma unroll
    for (int j = 0; j < 2; ++j) {
        ap[j] = lds + ((mh * 2 + j) * 16 + (lane & 15)) * LSTR + (laneHi << 3);
#pragma unroll
        for (int r = 0; r < 8; ++r) {
            int gr = m0 + (mh * 2 + j) * 16 + r + (laneHi << 3);
            int grc = gr < count ? gr : count - 1;
            int ent = bkt[grc];
            op[j][r]  = out + (size_t)(ent >> 1) * DDIM;
            rwv[j][r] = gr < count ? rweight[ent] : 0.0f;
        }
    }

    const __bf16* woe = wo + (size_t)e * DDIM * FDIM;

    for (int nd = ng; nd < DDIM / 16; nd += 4) {
        int n = nd * 16 + ncol;
        const __bf16* brow = woe + (size_t)n * FDIM + koff;
        v8f acc[2];
#pragma unroll
        for (int j = 0; j < 2; ++j) acc[j] = (v8f){0, 0, 0, 0, 0, 0, 0, 0};
        for (int k = 0; k < FDIM; k += 32) {
            v16bf b = load_frag_b(brow, k);
#pragma unroll
            for (int j = 0; j < 2; ++j)
                acc[j] = wmma_bf16(load_frag_a(ap[j], k), b, acc[j]);
        }
        float bias = bo[e * DDIM + n];
#pragma unroll
        for (int j = 0; j < 2; ++j) {
#pragma unroll
            for (int r = 0; r < 8; ++r)
                atomicAdd(op[j][r] + n, rwv[j][r] * (acc[j][r] + bias));
        }
    }
}

// ---------------------------------------------------------------------------
// Shared stage 2: out = sgate * (sinter@swo^T + sbo)   (initializes out)
// ---------------------------------------------------------------------------
__global__ __launch_bounds__(256) void shared_mlp2_kernel(
    const __bf16* __restrict__ sinter, const __bf16* __restrict__ swo,
    const float* __restrict__ sbo, const float* __restrict__ sgate,
    float* __restrict__ out) {
    extern __shared__ __bf16 lds[];
    const int LSTR = FDIM + LDS_PAD;
    int m0 = blockIdx.x * 64;

    {
        int tid = threadIdx.x;
        int lrow = tid >> 2;
        const v8bf* gp = (const v8bf*)(sinter + (size_t)(m0 + lrow) * FDIM + (tid & 3) * (FDIM / 4));
        v8bf* lp = (v8bf*)(lds + lrow * LSTR + (tid & 3) * (FDIM / 4));
#pragma unroll
        for (int c = 0; c < FDIM / 4 / 8; ++c) lp[c] = gp[c];
    }
    __syncthreads();

    int wave = threadIdx.x >> 5, lane = threadIdx.x & 31;
    int mh = wave & 1, ng = wave >> 1;
    int laneHi = lane >> 4;
    int koff = laneHi << 3;
    int ncol = lane & 15;

    const __bf16* ap[2];
    float* op[2][8];
    float  sgv[2][8];
#pragma unroll
    for (int j = 0; j < 2; ++j) {
        ap[j] = lds + ((mh * 2 + j) * 16 + (lane & 15)) * LSTR + (laneHi << 3);
#pragma unroll
        for (int r = 0; r < 8; ++r) {
            int m = m0 + (mh * 2 + j) * 16 + r + (laneHi << 3);
            op[j][r]  = out + (size_t)m * DDIM;
            sgv[j][r] = sgate[m];
        }
    }

    for (int nd = ng; nd < DDIM / 16; nd += 4) {
        int n = nd * 16 + ncol;
        const __bf16* brow = swo + (size_t)n * FDIM + koff;
        v8f acc[2];
#pragma unroll
        for (int j = 0; j < 2; ++j) acc[j] = (v8f){0, 0, 0, 0, 0, 0, 0, 0};
        for (int k = 0; k < FDIM; k += 32) {
            v16bf b = load_frag_b(brow, k);
#pragma unroll
            for (int j = 0; j < 2; ++j)
                acc[j] = wmma_bf16(load_frag_a(ap[j], k), b, acc[j]);
        }
        float bias = sbo[n];
#pragma unroll
        for (int j = 0; j < 2; ++j) {
#pragma unroll
            for (int r = 0; r < 8; ++r)
                op[j][r][n] = sgv[j][r] * (acc[j][r] + bias);
        }
    }
}

// ---------------------------------------------------------------------------
extern "C" void kernel_launch(void* const* d_in, const int* in_sizes, int n_in,
                              void* d_out, int out_size, void* d_ws, size_t ws_size,
                              hipStream_t stream) {
    (void)in_sizes; (void)n_in; (void)out_size; (void)ws_size;
    const float* h      = (const float*)d_in[0];
    const float* gate_w = (const float*)d_in[1];
    const float* w0     = (const float*)d_in[2];
    const float* b0     = (const float*)d_in[3];
    const float* w1     = (const float*)d_in[4];
    const float* b1     = (const float*)d_in[5];
    const float* wo     = (const float*)d_in[6];
    const float* bo     = (const float*)d_in[7];
    const float* sw0    = (const float*)d_in[8];
    const float* sb0    = (const float*)d_in[9];
    const float* sw1    = (const float*)d_in[10];
    const float* sb1    = (const float*)d_in[11];
    const float* swo    = (const float*)d_in[12];
    const float* sbo    = (const float*)d_in[13];
    const float* sg_w   = (const float*)d_in[14];

    float* out        = (float*)d_out;                       // final [T, D]
    float* logits_out = out + (size_t)T_TOK * DDIM;          // router_logits [T, E]

    // --- workspace layout (bytes) ---
    char* ws = (char*)d_ws;
    __bf16* h16     = (__bf16*)(ws);                         // 16,777,216
    __bf16* inter   = (__bf16*)(ws + 16777216);              //  8,388,608
    __bf16* sinter  = (__bf16*)(ws + 25165824);              //  4,194,304
    float*  sgate   = (float*) (ws + 29360128);              // T*4
    float*  rweight = (float*) (ws + 29376512);              // 2T*4
    int*    buckets = (int*)   (ws + 29409280);              // E*T*4
    int*    counts  = (int*)   (ws + 29540352);              // pad to 29540608
    __bf16* w0_16   = (__bf16*)(ws + 29540608);              // 16,777,216
    __bf16* w1_16   = (__bf16*)(ws + 46317824);              // 16,777,216
    __bf16* wo_16   = (__bf16*)(ws + 63095040);              // 16,777,216
    __bf16* sw0_16  = (__bf16*)(ws + 79872256);              //  2,097,152
    __bf16* sw1_16  = (__bf16*)(ws + 81969408);              //  2,097,152
    __bf16* swo_16  = (__bf16*)(ws + 84066560);              //  2,097,152

    const int SMEM1 = 64 * (DDIM + LDS_PAD) * 2;   // 263,168 B for stage-1 A tile
    const int SMEM2 = 64 * (FDIM + LDS_PAD) * 2;   //  66,560 B for stage-2 A tile
    hipFuncSetAttribute((const void*)mlp1_kernel,
                        hipFuncAttributeMaxDynamicSharedMemorySize, SMEM1);
    hipFuncSetAttribute((const void*)moe_mlp2_kernel,
                        hipFuncAttributeMaxDynamicSharedMemorySize, SMEM2);
    hipFuncSetAttribute((const void*)shared_mlp2_kernel,
                        hipFuncAttributeMaxDynamicSharedMemorySize, SMEM2);

    hipMemsetAsync(counts, 0, NEXP * sizeof(int), stream);

    // one-time bf16 casts: activations + all GEMM weights
    cast_bf16_kernel<<<(T_TOK * DDIM / 4) / 256, 256, 0, stream>>>(h, h16);
    cast_bf16_kernel<<<(NEXP * FDIM * DDIM / 4) / 256, 256, 0, stream>>>(w0, w0_16);
    cast_bf16_kernel<<<(NEXP * FDIM * DDIM / 4) / 256, 256, 0, stream>>>(w1, w1_16);
    cast_bf16_kernel<<<(NEXP * DDIM * FDIM / 4) / 256, 256, 0, stream>>>(wo, wo_16);
    cast_bf16_kernel<<<(FDIM * DDIM / 4) / 256, 256, 0, stream>>>(sw0, sw0_16);
    cast_bf16_kernel<<<(FDIM * DDIM / 4) / 256, 256, 0, stream>>>(sw1, sw1_16);
    cast_bf16_kernel<<<(DDIM * FDIM / 4) / 256, 256, 0, stream>>>(swo, swo_16);

    router_kernel<<<T_TOK / 8, 256, 0, stream>>>(h, gate_w, sg_w, logits_out,
                                                 sgate, rweight, buckets, counts);

    // shared expert (identity rows)
    mlp1_kernel<<<dim3(T_TOK / 64, 1), 256, SMEM1, stream>>>(
        h16, sw0_16, sb0, sw1_16, sb1, nullptr, nullptr, sinter);
    shared_mlp2_kernel<<<T_TOK / 64, 256, SMEM2, stream>>>(sinter, swo_16, sbo, sgate, out);

    // routed experts (gathered rows)
    dim3 egrid(T_TOK / 64, NEXP);
    mlp1_kernel<<<egrid, 256, SMEM1, stream>>>(
        h16, w0_16, b0, w1_16, b1, buckets, counts, inter);
    moe_mlp2_kernel<<<egrid, 256, SMEM2, stream>>>(
        inter, wo_16, bo, buckets, counts, rweight, out);
}
